// GuidedSampler_53996328845466
// MI455X (gfx1250) — matrix operand order repaired and checked by reading, not compiled, not run on hardware
//
#include <hip/hip_runtime.h>

// GuidedSampler for MI455X (gfx1250, wave32, WMMA).
// Pass 1: fused W[512x64] x F[64x4096] GEMM per batch via v_wmma_f32_16x16x32_bf16
//         using split-bf16 (hi/lo) emulated-f32 products; squared-distance
//         reduction fused into the WMMA epilogue (kv never materialized).
//         Waves split the 512-row M dimension (4 M-tiles each); the 64ch x 16px
//         feature tile is cooperatively staged in LDS in WMMA-B layout each step,
//         so per-wave register state is tiny and nothing spills.
// Pass 2: argmin -> codes; selected 1x1 conv recomputed in exact f32 + MSE loss.

typedef __attribute__((ext_vector_type(16))) __bf16 v16bf;
typedef __attribute__((ext_vector_type(8)))  float  v8f;

#define B_    64
#define DIM_  64
#define DQ_   8
#define HW_   4096
#define K_    64
#define M_    512          // K_*DQ_
#define MT_   32           // M tiles of 16
#define SEG_  4            // pixel segments per batch (blocks per batch)
#define JT_   64           // 16-pixel tiles per segment (4096/16/SEG_)

__device__ __forceinline__ __bf16 f2bf(float x) {
    unsigned u = __builtin_bit_cast(unsigned, x);
    unsigned r = u + 0x7FFFu + ((u >> 16) & 1u);      // RNE
    unsigned short h = (unsigned short)(r >> 16);
    return __builtin_bit_cast(__bf16, h);
}
__device__ __forceinline__ float bf2f(__bf16 h) {
    unsigned u = ((unsigned)__builtin_bit_cast(unsigned short, h)) << 16;
    return __builtin_bit_cast(float, u);
}

// ---------------------------------------------------------------------------
// Kernel 1: fused distance^2 partials.  grid = (SEG_, B_), block = 256.
// Wave w owns M-tiles [4w, 4w+4)  ->  codes k in [8w, 8w+8).
// ---------------------------------------------------------------------------
__global__ __launch_bounds__(256) void gs_dist_kernel(
    const float* __restrict__ feat,    // [B,64,4096]
    const float* __restrict__ query,   // [B,8,4096]
    const float* __restrict__ weight,  // [64,8,64] == [512,64]
    float* __restrict__ part_ws)       // [B*SEG_*K_] per-block partials
{
    // A fragments pre-swizzled: frag f = mt*4 + kc*2 + term (term 0=hi,1=lo)
    // each frag: 32 lanes * 16 bf16 (32B contiguous per lane) = 1024B
    __shared__ __attribute__((aligned(32))) __bf16 aLds[MT_ * 4 * 512]; // 128KB
    // B fragments for the current 16-pixel tile: frag g = term*2 + kc
    __shared__ __attribute__((aligned(32))) __bf16 bLds[4 * 512];       // 4KB

    const int b    = blockIdx.y;
    const int seg  = blockIdx.x;
    const int tid  = threadIdx.x;
    const int lane = tid & 31;
    const int wave = tid >> 5;
    const int col  = lane & 15;
    const int half = lane >> 4;
    const int mt0  = wave * 4;

    // ---- stage swizzled hi/lo weight fragments into LDS (once) ----
    for (int slot = tid; slot < 128 * 32; slot += 256) {
        const int f    = slot >> 5;
        const int l    = slot & 31;
        const int mt   = f >> 2;
        const int kc   = (f >> 1) & 1;
        const int term = f & 1;
        const int m    = mt * 16 + (l & 15);
        const int lh   = l >> 4;
        __bf16* dst = &aLds[f * 512 + l * 16];
        #pragma unroll
        for (int i = 0; i < 16; ++i) {
            // A-matrix 16x32: contiguous k order, k = lh*16 + i (matches B staging)
            const int v  = i >> 1;
            const int kl = lh * 16 + i;
            const int c  = kc * 32 + kl;
            const float x = weight[m * DIM_ + c];
            const __bf16 hi = f2bf(x);
            dst[i] = term ? f2bf(x - bf2f(hi)) : hi;
        }
    }

    const float* fb = feat  + (size_t)b * DIM_ * HW_;
    const float* qb = query + (size_t)b * DQ_  * HW_;

    float ksum[4] = {0.f, 0.f, 0.f, 0.f};

    __syncthreads();

    for (int j = 0; j < JT_; ++j) {
        const int p0 = (seg * JT_ + j) * 16;

        // ---- cooperative B staging: 64ch x 16px tile -> hi/lo WMMA-B frags ----
        // B 32x16 layout: lane = halfB*16 + px, element i -> c = kc*32+halfB*16+i
        for (int t = tid; t < DIM_ * 16; t += 256) {
            const int c = t >> 4;
            const int p = t & 15;
            const float x = fb[c * HW_ + p0 + p];
            __builtin_prefetch(&fb[c * HW_ + p0 + p + 16], 0, 0); // next tile
            const __bf16 hi = f2bf(x);
            const __bf16 lo = f2bf(x - bf2f(hi));
            const int kc = c >> 5;
            const int hb = (c >> 4) & 1;
            const int i  = c & 15;
            const int ln = hb * 16 + p;
            bLds[(kc * 32 + ln) * 16 + i]       = hi;   // frags 0,1: hi
            bLds[((2 + kc) * 32 + ln) * 16 + i] = lo;   // frags 2,3: lo
        }
        __syncthreads();

        const v16bf bhi0 = *(const v16bf*)&bLds[(0 * 32 + lane) * 16];
        const v16bf bhi1 = *(const v16bf*)&bLds[(1 * 32 + lane) * 16];
        const v16bf blo0 = *(const v16bf*)&bLds[(2 * 32 + lane) * 16];
        const v16bf blo1 = *(const v16bf*)&bLds[(3 * 32 + lane) * 16];

        // query values for this column: C-frag VGPR r maps to q = r
        float qv[DQ_];
        #pragma unroll
        for (int r = 0; r < DQ_; ++r) qv[r] = qb[r * HW_ + p0 + col];

        #pragma unroll
        for (int u = 0; u < 4; ++u) {
            const int mt = mt0 + u;
            const __bf16* ab = &aLds[mt * 4 * 512 + lane * 16];
            const v16bf ahi0 = *(const v16bf*)(ab);
            const v16bf alo0 = *(const v16bf*)(ab + 512);
            const v16bf ahi1 = *(const v16bf*)(ab + 1024);
            const v16bf alo1 = *(const v16bf*)(ab + 1536);

            v8f acc = {};
            // emulated-f32: hi*hi + hi*lo + lo*hi, f32 accumulate
            acc = __builtin_amdgcn_wmma_f32_16x16x32_bf16(false, ahi0, false, bhi0, (short)0, acc, false, false);
            acc = __builtin_amdgcn_wmma_f32_16x16x32_bf16(false, ahi0, false, blo0, (short)0, acc, false, false);
            acc = __builtin_amdgcn_wmma_f32_16x16x32_bf16(false, alo0, false, bhi0, (short)0, acc, false, false);
            acc = __builtin_amdgcn_wmma_f32_16x16x32_bf16(false, ahi1, false, bhi1, (short)0, acc, false, false);
            acc = __builtin_amdgcn_wmma_f32_16x16x32_bf16(false, ahi1, false, blo1, (short)0, acc, false, false);
            acc = __builtin_amdgcn_wmma_f32_16x16x32_bf16(false, alo1, false, bhi1, (short)0, acc, false, false);

            // epilogue: this lane's rows m = mt*16 + half*8 + r  ->  q=r, k=2*mt+half
            float part = 0.f;
            #pragma unroll
            for (int r = 0; r < DQ_; ++r) {
                const float d = acc[r] - qv[r];
                part = fmaf(d, d, part);
            }
            ksum[u] += part;
        }
        __syncthreads();   // before next iteration overwrites bLds
    }

    // reduce the 16 columns within each half; each k owned by one wave/half
    #pragma unroll
    for (int u = 0; u < 4; ++u) {
        float v = ksum[u];
        v += __shfl_xor(v, 1, 32);
        v += __shfl_xor(v, 2, 32);
        v += __shfl_xor(v, 4, 32);
        v += __shfl_xor(v, 8, 32);
        if (col == 0) {
            const int k = (mt0 + u) * 2 + half;
            part_ws[((size_t)b * SEG_ + seg) * K_ + k] = v;
        }
    }
}

// ---------------------------------------------------------------------------
// Kernel 2: sum segment partials (fixed order) + argmin -> codes
// ---------------------------------------------------------------------------
__global__ void gs_argmin_kernel(const float* __restrict__ part_ws,
                                 int* __restrict__ codes_ws,
                                 float* __restrict__ out_codes)
{
    const int b = threadIdx.x;
    if (b >= B_) return;
    float best = 3.4e38f;
    int   bi   = 0;
    for (int k = 0; k < K_; ++k) {
        float s = 0.f;
        #pragma unroll
        for (int seg = 0; seg < SEG_; ++seg)
            s += part_ws[((size_t)b * SEG_ + seg) * K_ + k];
        if (s < best) { best = s; bi = k; }   // first-min tie-break == argmax(-dist)
    }
    codes_ws[b] = bi;
    out_codes[b] = (float)bi;
}

// ---------------------------------------------------------------------------
// Kernel 3: selected 1x1 conv (exact f32) + per-block loss partial
// ---------------------------------------------------------------------------
__global__ __launch_bounds__(256) void gs_select_kernel(
    const float* __restrict__ feat, const float* __restrict__ query,
    const float* __restrict__ weight, const int* __restrict__ codes_ws,
    float* __restrict__ out_sel, float* __restrict__ loss_ws)
{
    __shared__ float wsel[DQ_ * DIM_];
    __shared__ float red[8];
    const int b   = blockIdx.x;
    const int tid = threadIdx.x;
    const int code = codes_ws[b];
    for (int t = tid; t < DQ_ * DIM_; t += 256)
        wsel[t] = weight[(size_t)code * DQ_ * DIM_ + t];
    __syncthreads();

    const float* fb = feat  + (size_t)b * DIM_ * HW_;
    const float* qb = query + (size_t)b * DQ_  * HW_;
    float*       ob = out_sel + (size_t)b * DQ_ * HW_;

    float lsum = 0.f;
    for (int p = tid; p < HW_; p += 256) {
        float acc[DQ_] = {};
        #pragma unroll 8
        for (int c = 0; c < DIM_; ++c) {
            const float f = fb[c * HW_ + p];
            #pragma unroll
            for (int r = 0; r < DQ_; ++r) acc[r] = fmaf(wsel[r * DIM_ + c], f, acc[r]);
        }
        #pragma unroll
        for (int r = 0; r < DQ_; ++r) {
            ob[r * HW_ + p] = acc[r];
            const float d = acc[r] - qb[r * HW_ + p];
            lsum = fmaf(d, d, lsum);
        }
    }
    lsum += __shfl_xor(lsum, 1, 32);
    lsum += __shfl_xor(lsum, 2, 32);
    lsum += __shfl_xor(lsum, 4, 32);
    lsum += __shfl_xor(lsum, 8, 32);
    lsum += __shfl_xor(lsum, 16, 32);
    if ((tid & 31) == 0) red[tid >> 5] = lsum;
    __syncthreads();
    if (tid == 0) {
        float s = 0.f;
        #pragma unroll
        for (int w = 0; w < 8; ++w) s += red[w];
        loss_ws[b] = s;                // per-block slot: deterministic
    }
}

// ---------------------------------------------------------------------------
// Kernel 4: finalize loss (fixed-order sum over 64 block partials)
// ---------------------------------------------------------------------------
__global__ void gs_loss_kernel(const float* __restrict__ loss_ws,
                               float* __restrict__ out_loss)
{
    if (threadIdx.x == 0) {
        float s = 0.f;
        for (int b = 0; b < B_; ++b) s += loss_ws[b];
        out_loss[0] = s / (float)((size_t)B_ * DQ_ * HW_);
    }
}

extern "C" void kernel_launch(void* const* d_in, const int* in_sizes, int n_in,
                              void* d_out, int out_size, void* d_ws, size_t ws_size,
                              hipStream_t stream) {
    (void)in_sizes; (void)n_in; (void)out_size; (void)ws_size;
    const float* feat   = (const float*)d_in[0];  // [64,64,64,64]
    const float* query  = (const float*)d_in[1];  // [64,8,64,64]
    const float* weight = (const float*)d_in[2];  // [64,8,64]

    float* out       = (float*)d_out;
    float* out_sel   = out;                               // 2,097,152
    float* out_codes = out + (size_t)B_ * DQ_ * HW_;      // 64
    float* out_loss  = out_codes + B_;                    // 1

    float* part_ws  = (float*)d_ws;                       // B*SEG_*K_ = 16384 f
    int*   codes_ws = (int*)(part_ws + (size_t)B_ * SEG_ * K_);
    float* loss_ws  = (float*)(codes_ws + B_);            // 64 f

    dim3 grid(SEG_, B_);
    gs_dist_kernel<<<grid, 256, 0, stream>>>(feat, query, weight, part_ws);
    gs_argmin_kernel<<<1, 64, 0, stream>>>(part_ws, codes_ws, out_codes);
    gs_select_kernel<<<B_, 256, 0, stream>>>(feat, query, weight, codes_ws,
                                             out_sel, loss_ws);
    gs_loss_kernel<<<1, 1, 0, stream>>>(loss_ws, out_loss);
}
